// DeformableConv_20538533610096
// MI455X (gfx1250) — matrix operand order repaired and checked by reading, not compiled, not run on hardware
//
#include <hip/hip_runtime.h>
#include <hip/hip_bf16.h>

typedef __attribute__((ext_vector_type(16))) _Float16 v16h;
typedef __attribute__((ext_vector_type(8)))  _Float16 v8h;
typedef __attribute__((ext_vector_type(8)))  float    v8f;

#define BATCH 8
#define CIN   64
#define HH    128
#define WW    128
#define OUTC  64
#define NTAP  9
#define KTOT  576   // CIN * NTAP, tap-major: k = tap*64 + c

// ---------------------------------------------------------------------------
// Kernel 0: repack w_deform (outC, C, 3, 3) f32 -> A[o][tap*64 + c] f16
// ---------------------------------------------------------------------------
__global__ __launch_bounds__(256) void pack_weights(const float* __restrict__ w_def,
                                                    _Float16* __restrict__ A) {
    int i = blockIdx.x * 256 + threadIdx.x;
    if (i >= OUTC * KTOT) return;
    int o   = i / KTOT;
    int r   = i % KTOT;
    int tap = r >> 6;          // r / 64
    int c   = r & 63;          // r % 64
    // src flat (o, c, kh, kw) = o*576 + c*9 + tap
    A[i] = (_Float16)w_def[(size_t)o * KTOT + c * NTAP + tap];
}

// ---------------------------------------------------------------------------
// Kernel 1: offset = conv3x3(x, w_offset), 18 output channels.
// One workgroup per (b, h). LDS slab of 32 channels x 3 rows, two passes.
// ---------------------------------------------------------------------------
__global__ __launch_bounds__(256) void offset_conv(const float* __restrict__ x,
                                                   const float* __restrict__ w_off,
                                                   float* __restrict__ offs) {
    __shared__ float slab[32 * 3 * 128];   // 48 KB
    const int tid = threadIdx.x;
    const int wg  = blockIdx.x;
    const int b   = wg >> 7;
    const int h   = wg & 127;

    const int wcol = tid & 127;
    const int grp  = tid >> 7;             // 0..1 -> channels grp*9 .. grp*9+8

    float acc[9];
#pragma unroll
    for (int i = 0; i < 9; ++i) acc[i] = 0.f;

    for (int cb = 0; cb < 2; ++cb) {
        // stage 32 channels x rows (h-1..h+1), zero-padded
        for (int i = tid; i < 32 * 3 * 128; i += 256) {
            int c_l = i / 384;
            int rem = i % 384;
            int r   = rem >> 7;
            int w   = rem & 127;
            int y   = h + r - 1;
            float v = 0.f;
            if (y >= 0 && y < HH)
                v = x[(((size_t)b * CIN + cb * 32 + c_l) * HH + y) * WW + w];
            slab[i] = v;
        }
        __syncthreads();

        for (int oi = 0; oi < 9; ++oi) {
            int ch = grp * 9 + oi;
            float a = acc[oi];
            for (int c_l = 0; c_l < 32; ++c_l) {
                const float* wt = w_off + ((size_t)ch * CIN + cb * 32 + c_l) * 9;
                const float* xr = slab + c_l * 384;
#pragma unroll
                for (int kh = 0; kh < 3; ++kh) {
#pragma unroll
                    for (int kw = 0; kw < 3; ++kw) {
                        int wx = wcol + kw - 1;
                        float xv = (wx >= 0 && wx < WW) ? xr[kh * 128 + wx] : 0.f;
                        a += xv * wt[kh * 3 + kw];
                    }
                }
            }
            acc[oi] = a;
        }
        __syncthreads();
    }
#pragma unroll
    for (int oi = 0; oi < 9; ++oi) {
        int ch = grp * 9 + oi;
        offs[(((size_t)b * 18 + ch) * HH + h) * WW + wcol] = acc[oi];
    }
}

// ---------------------------------------------------------------------------
// Kernel 2: fused bilinear gather + GEMM via v_wmma_f32_16x16x32_f16.
// One workgroup (8 waves) per (b, h): computes out[b, 0:64, h, 0:128].
// K loop: 18 steps of (tap, c_half); cols tile 32(K) x 128(N) f16 in LDS.
// Wave (m,n) tiling: mtile = wave&3 (16 rows), ntile = wave>>2 (64 cols).
// ---------------------------------------------------------------------------
__global__ __launch_bounds__(256) void deform_gemm(const float* __restrict__ x,
                                                   const float* __restrict__ offs,
                                                   const _Float16* __restrict__ A,
                                                   float* __restrict__ out) {
    // cols[n][k]: per-pixel 32 contiguous K halves -> one aligned 32B lane read
    __shared__ _Float16 s_cols[128 * 32] __attribute__((aligned(64)));   // 8 KB
    __shared__ int   s_i00[NTAP * 128], s_i01[NTAP * 128];
    __shared__ int   s_i10[NTAP * 128], s_i11[NTAP * 128];               // 18 KB
    __shared__ float s_w00[NTAP * 128], s_w01[NTAP * 128];
    __shared__ float s_w10[NTAP * 128], s_w11[NTAP * 128];               // 18 KB

    const int tid  = threadIdx.x;
    const int wg   = blockIdx.x;
    const int b    = wg >> 7;
    const int h    = wg & 127;
    const int lane = tid & 31;
    const int wave = tid >> 5;
    const int mtile = wave & 3;     // 4 tiles of 16 output channels
    const int ntile = wave >> 2;    // 2 tiles of 64 pixels

    // ---- precompute bilinear corners/weights for 9 taps x 128 pixels ----
    for (int e = tid; e < NTAP * 128; e += 256) {
        int tap = e >> 7;
        int n   = e & 127;
        float offy = offs[(((size_t)b * 18 + tap * 2    ) * HH + h) * WW + n];
        float offx = offs[(((size_t)b * 18 + tap * 2 + 1) * HH + h) * WW + n];
        float py = (float)h + (float)(tap / 3 - 1) + offy;
        float px = (float)n + (float)(tap % 3 - 1) + offx;
        float y0f = floorf(py), x0f = floorf(px);
        int y0 = (int)y0f, x0 = (int)x0f;
        int y1 = y0 + 1,   x1 = x0 + 1;
        float wy1 = py - y0f, wx1 = px - x0f;
        float wy0 = 1.f - wy1, wx0 = 1.f - wx1;
        float vy0 = (y0 >= 0 && y0 < HH) ? 1.f : 0.f;
        float vy1 = (y1 >= 0 && y1 < HH) ? 1.f : 0.f;
        float vx0 = (x0 >= 0 && x0 < WW) ? 1.f : 0.f;
        float vx1 = (x1 >= 0 && x1 < WW) ? 1.f : 0.f;
        int y0c = min(max(y0, 0), HH - 1), y1c = min(max(y1, 0), HH - 1);
        int x0c = min(max(x0, 0), WW - 1), x1c = min(max(x1, 0), WW - 1);
        s_i00[e] = y0c * WW + x0c;  s_i01[e] = y0c * WW + x1c;
        s_i10[e] = y1c * WW + x0c;  s_i11[e] = y1c * WW + x1c;
        s_w00[e] = wy0 * wx0 * vy0 * vx0;  s_w01[e] = wy0 * wx1 * vy0 * vx1;
        s_w10[e] = wy1 * wx0 * vy1 * vx0;  s_w11[e] = wy1 * wx1 * vy1 * vx1;
    }

    v8f acc[4];
#pragma unroll
    for (int t = 0; t < 4; ++t) acc[t] = (v8f){};

    const int half = tid >> 7;      // gather role: which 16-channel group
    const int npix = tid & 127;     // gather role: pixel in row

    // ---- K loop: 18 steps of (tap, c_half) ----
    for (int step = 0; step < 18; ++step) {
        const int tap   = step >> 1;
        const int chalf = step & 1;

        __syncthreads();   // prev WMMAs done reading s_cols (and coords ready)

        // gather: 32 channels x 128 pixels of bilinear samples -> f16 LDS
        {
            int e = tap * 128 + npix;
            int i00 = s_i00[e], i01 = s_i01[e], i10 = s_i10[e], i11 = s_i11[e];
            float w00 = s_w00[e], w01 = s_w01[e], w10 = s_w10[e], w11 = s_w11[e];
            int cbase = chalf * 32 + half * 16;
#pragma unroll
            for (int i = 0; i < 16; ++i) {
                const float* xp = x + ((size_t)b * CIN + cbase + i) * (HH * WW);
                float v = w00 * xp[i00] + w01 * xp[i01] +
                          w10 * xp[i10] + w11 * xp[i11];
                s_cols[npix * 32 + half * 16 + i] = (_Float16)v;
            }
        }
        __syncthreads();

        // A fragment (this code's contiguous k order, same as B): lanes 0-15 take K 0-15,
        // lanes 16-31 take K 16-31 of this step's 32-wide K window.
        v16h af;
        {
            int o   = mtile * 16 + (lane & 15);
            int sel = (lane >> 4) * 16;
            const _Float16* ap = A + (size_t)o * KTOT + tap * 64 + chalf * 32;
            v8h lo = *(const v8h*)(ap + sel);
            v8h hi = *(const v8h*)(ap + 8 + sel);
#pragma unroll
            for (int i = 0; i < 8; ++i) { af[i] = lo[i]; af[i + 8] = hi[i]; }
        }

#pragma unroll
        for (int t = 0; t < 4; ++t) {
            int n_loc  = ntile * 64 + t * 16 + (lane & 15);
            int kbase  = (lane >> 4) * 16;   // B 32x16: lanes 0-15 K0-15, 16-31 K16-31
            v16h bf = *(const v16h*)(s_cols + n_loc * 32 + kbase);
            acc[t] = __builtin_amdgcn_wmma_f32_16x16x32_f16(
                false, af, false, bf, (short)0, acc[t], false, false);
        }
    }

    // ---- epilogue: D layout (m = (lane>>4)*8 + j, n = lane&15) ----
#pragma unroll
    for (int t = 0; t < 4; ++t) {
#pragma unroll
        for (int j = 0; j < 8; ++j) {
            int o    = mtile * 16 + (lane >> 4) * 8 + j;
            int wcol = ntile * 64 + t * 16 + (lane & 15);
            out[(((size_t)b * OUTC + o) * HH + h) * WW + wcol] = acc[t][j];
        }
    }
}

// ---------------------------------------------------------------------------
extern "C" void kernel_launch(void* const* d_in, const int* in_sizes, int n_in,
                              void* d_out, int out_size, void* d_ws, size_t ws_size,
                              hipStream_t stream) {
    const float* x     = (const float*)d_in[0];   // (8, 64, 128, 128)
    const float* w_off = (const float*)d_in[1];   // (18, 64, 3, 3)
    const float* w_def = (const float*)d_in[2];   // (64, 64, 3, 3)
    float* out = (float*)d_out;                   // (8, 64, 128, 128)

    const size_t offs_bytes = (size_t)BATCH * 18 * HH * WW * sizeof(float); // 9.44 MB
    float*     offs = (float*)d_ws;
    _Float16*  Af16 = (_Float16*)((char*)d_ws + offs_bytes);                // 72 KB

    pack_weights<<<(OUTC * KTOT + 255) / 256, 256, 0, stream>>>(w_def, Af16);
    offset_conv<<<BATCH * HH, 256, 0, stream>>>(x, w_off, offs);
    deform_gemm<<<BATCH * HH, 256, 0, stream>>>(x, offs, Af16, out);
}
